// HeteroTemporalGNN_47193100648624
// MI455X (gfx1250) — matrix operand order repaired and checked by reading, not compiled, not run on hardware
//
#include <hip/hip_runtime.h>
#include <hip/hip_bf16.h>

typedef __attribute__((ext_vector_type(16))) _Float16 v16h;
typedef __attribute__((ext_vector_type(8)))  _Float16 h8;
typedef __attribute__((ext_vector_type(8)))  float    v8f;

namespace {
constexpr int kNT = 50000, kNV = 1000, kNDg = 20000, kNM = 5000, kNP = 8000;
constexpr int kE = 100000, kD = 256, kH = 4, kC = 64, kL = 2, kT = 8;
constexpr int kNTOT = kNT + kNV + kNDg + kNM + kNP;   // 84000
constexpr int kNEMB = kNV + kNDg + kNM + kNP;         // 34000
}

// ---------------------------------------------------------------------------
// WMMA GEMM, f16 operands / f32 accum.  A[M,K] row-major f16.  Wt[N,K] f16
// (W pre-transposed).  N fixed at 256.  Block (32,4): the 16-row A tile is
// staged once into LDS via async global->LDS B128 copies (it is contiguous in
// memory); each of the 4 waves then computes a 16x64 strip (4 accumulators,
// LDS A fragment reused 4x per k-step, B streamed from global as 2x b128).
// K in {64,128,256} -> fully unrolled: 4*K/32 straight-line v_wmma per block.
// ---------------------------------------------------------------------------
template <int K>
__global__ void gemm_wmma_h(const _Float16* __restrict__ A,
                            const _Float16* __restrict__ Wt,
                            const float* __restrict__ bias,
                            float* __restrict__ Out, int M)
{
  constexpr int N = 256;
  __shared__ _Float16 sA[16 * K];                 // 16-row A tile (<= 8KB)

  const int lane = threadIdx.x;                   // 0..31
  const int ml   = lane & 15;
  const int hsw  = lane >> 4;
  const int m0   = blockIdx.x * 16;
  const int n0w  = threadIdx.y * 64;              // wave's 64-column strip
  const int tid  = threadIdx.y * 32 + lane;       // 0..127

  // ---- async stage of contiguous A tile: 32*K bytes, 16B per thread/round ----
  {
    const _Float16* Atile = A + (size_t)m0 * K;   // rows m0..m0+15 contiguous
    constexpr int rounds = (16 * K * 2) / (128 * 16);   // 1 / 2 / 4
    unsigned lbase = (unsigned)(uintptr_t)(&sA[0]);
#pragma unroll
    for (int r = 0; r < rounds; ++r) {
      unsigned    loff = lbase + (unsigned)((r * 128 + tid) * 16);
      const void* gptr = (const char*)Atile + (size_t)(r * 128 + tid) * 16;
      asm volatile("global_load_async_to_lds_b128 %0, %1, off"
                   :: "v"(loff), "v"(gptr) : "memory");
    }
    asm volatile("s_wait_asynccnt 0x0" ::: "memory");
    __syncthreads();
  }

  v8f c0 = {}, c1 = {}, c2 = {}, c3 = {};
#pragma unroll
  for (int k0 = 0; k0 < K; k0 += 32) {
    // A fragment from LDS: row ml; K = k0+16*hsw+j (contiguous)
    const _Float16* ar = &sA[ml * K];
    h8 alo = *(const h8*)(ar + k0 + (hsw << 4));
    h8 ahi = *(const h8*)(ar + k0 + (hsw << 4) + 8);
    v16h a = __builtin_shufflevector(alo, ahi, 0,1,2,3,4,5,6,7,8,9,10,11,12,13,14,15);
    // B fragment (tile t): column n = n0w+16t+ml ; K = k0+16*hsw+j (contiguous)
#pragma unroll
    for (int t = 0; t < 4; ++t) {
      const _Float16* bp = Wt + (size_t)(n0w + 16 * t + ml) * K + k0 + (hsw << 4);
      h8 blo = *(const h8*)(bp);
      h8 bhi = *(const h8*)(bp + 8);
      v16h b = __builtin_shufflevector(blo, bhi, 0,1,2,3,4,5,6,7,8,9,10,11,12,13,14,15);
      if      (t == 0) c0 = __builtin_amdgcn_wmma_f32_16x16x32_f16(false,a,false,b,(short)0,c0,false,false);
      else if (t == 1) c1 = __builtin_amdgcn_wmma_f32_16x16x32_f16(false,a,false,b,(short)0,c1,false,false);
      else if (t == 2) c2 = __builtin_amdgcn_wmma_f32_16x16x32_f16(false,a,false,b,(short)0,c2,false,false);
      else             c3 = __builtin_amdgcn_wmma_f32_16x16x32_f16(false,a,false,b,(short)0,c3,false,false);
    }
  }

  // C/D layout: col N = lane%16 ; VGPR r -> row M = r + 8*hsw
  const int rbase = m0 + (hsw << 3);
  if (m0 + 16 <= M) {                                   // fast path: full tile
#pragma unroll
    for (int t = 0; t < 4; ++t) {
      const v8f& c = (t == 0) ? c0 : (t == 1) ? c1 : (t == 2) ? c2 : c3;
      const int n = n0w + 16 * t + ml;
      const float bv = bias ? bias[n] : 0.0f;
#pragma unroll
      for (int r = 0; r < 8; ++r)
        Out[(size_t)(rbase + r) * N + n] = c[r] + bv;
    }
  } else {                                              // tail tile
#pragma unroll
    for (int t = 0; t < 4; ++t) {
      const v8f& c = (t == 0) ? c0 : (t == 1) ? c1 : (t == 2) ? c2 : c3;
      const int n = n0w + 16 * t + ml;
      const float bv = bias ? bias[n] : 0.0f;
#pragma unroll
      for (int r = 0; r < 8; ++r)
        if (rbase + r < M) Out[(size_t)(rbase + r) * N + n] = c[r] + bv;
    }
  }
}

// ---- small helper kernels --------------------------------------------------
__global__ void k_cvt_h(const float* __restrict__ in, _Float16* __restrict__ out, int n)
{
  int i = blockIdx.x * blockDim.x + threadIdx.x;
  if (i < n) out[i] = (_Float16)in[i];
}

// Wt[n,k] = (f16) W[k,n]
__global__ void k_transpose_h(const float* __restrict__ W, _Float16* __restrict__ Wt,
                              int K, int N)
{
  int i = blockIdx.x * blockDim.x + threadIdx.x;
  if (i >= K * N) return;
  int k = i / N, n = i - k * N;
  Wt[(size_t)n * K + k] = (_Float16)W[i];
}

// wdeff[k,h] = sum_c Wd[k, h*64+c] * ad[h,c]
__global__ void k_wdeff(const float* __restrict__ Wd, const float* __restrict__ ad,
                        float* __restrict__ wdeff)
{
  int i = blockIdx.x * blockDim.x + threadIdx.x;
  if (i >= kD * kH) return;
  int k = i >> 2, h = i & 3;
  float s = 0.f;
  for (int c = 0; c < kC; ++c) s += Wd[k * kD + h * kC + c] * ad[h * kC + c];
  wdeff[k * kH + h] = s;
}

__global__ void k_ed(const float* __restrict__ xd, const float* __restrict__ wdeff,
                     float* __restrict__ ed, int Nd)
{
  int i = blockIdx.x * blockDim.x + threadIdx.x;
  if (i >= Nd * kH) return;
  int n = i >> 2, h = i & 3;
  float s = 0.f;
  for (int k = 0; k < kD; ++k) s += xd[(size_t)n * kD + k] * wdeff[k * kH + h];
  ed[i] = s;
}

__global__ void k_es(const float* __restrict__ hsb, const float* __restrict__ as,
                     float* __restrict__ es, int Ns)
{
  int i = blockIdx.x * blockDim.x + threadIdx.x;
  if (i >= Ns * kH) return;
  int n = i >> 2, h = i & 3;
  float s = 0.f;
  for (int c = 0; c < kC; ++c) s += hsb[(size_t)n * kD + h * kC + c] * as[h * kC + c];
  es[i] = s;
}

__global__ void k_initmd(float* __restrict__ m, float* __restrict__ den, int Nd)
{
  int i = blockIdx.x * blockDim.x + threadIdx.x;
  if (i < Nd * kH) { m[i] = -1e30f; den[i] = 0.f; }
}

__device__ inline void atomicMaxF(float* addr, float val)
{
  int old = __float_as_int(*addr);
  while (__int_as_float(old) < val) {
    int prev = atomicCAS((int*)addr, old, __float_as_int(val));
    if (prev == old) break;
    old = prev;
  }
}

__device__ inline float edge_logit(const int* ei, const float* es, const float* ed,
                                   int e, int h)
{
  int s = ei[e], d = ei[kE + e];
  float v = es[s * kH + h] + ed[d * kH + h];
  return v > 0.f ? v : 0.2f * v;                      // leaky relu 0.2
}

__global__ void k_edgemax(const int* __restrict__ ei, const float* __restrict__ es,
                          const float* __restrict__ ed, float* __restrict__ m)
{
  int i = blockIdx.x * blockDim.x + threadIdx.x;
  if (i >= kE * kH) return;
  int e = i >> 2, h = i & 3;
  int d = ei[kE + e];
  atomicMaxF(&m[d * kH + h], edge_logit(ei, es, ed, e, h));
}

__global__ void k_edgeexp(const int* __restrict__ ei, const float* __restrict__ es,
                          const float* __restrict__ ed, const float* __restrict__ m,
                          float* __restrict__ den, float* __restrict__ exb)
{
  int i = blockIdx.x * blockDim.x + threadIdx.x;
  if (i >= kE * kH) return;
  int e = i >> 2, h = i & 3;
  int d = ei[kE + e];
  float ex = expf(edge_logit(ei, es, ed, e, h) - m[d * kH + h]);
  exb[i] = ex;
  atomicAdd(&den[d * kH + h], ex);
}

__global__ void k_scatter(const int* __restrict__ ei, const float* __restrict__ hsb,
                          const float* __restrict__ exb, const float* __restrict__ den,
                          float* __restrict__ outd)
{
  int e = blockIdx.x, j = threadIdx.x, h = j >> 6;
  int s = ei[e], d = ei[kE + e];
  float alpha = exb[e * kH + h] / (den[d * kH + h] + 1e-16f);
  atomicAdd(&outd[(size_t)d * kD + j], alpha * hsb[(size_t)s * kD + j]);
}

__global__ void k_addbias(const float* __restrict__ b, float* __restrict__ outd, int Nd)
{
  int i = blockIdx.x * blockDim.x + threadIdx.x;
  if (i < Nd * kD) outd[i] += b[i & (kD - 1)];
}

// relu into f32 (for next-layer GEMV/mean) and f16 (for next-layer WMMA GEMMs)
__global__ void k_relu_cvt(const float* __restrict__ xin, float* __restrict__ xout,
                           _Float16* __restrict__ xh, int n)
{
  int i = blockIdx.x * blockDim.x + threadIdx.x;
  if (i < n) {
    float v = xin[i]; v = v > 0.f ? v : 0.f;
    xout[i] = v; xh[i] = (_Float16)v;
  }
}

__global__ void k_mean_ts(const float* __restrict__ x, float* __restrict__ g)
{
  __shared__ float sm[256];
  int col = blockIdx.x;
  float s = 0.f;
  for (int i = threadIdx.x; i < kNT; i += 256) s += x[(size_t)i * kD + col];
  sm[threadIdx.x] = s;
  __syncthreads();
  for (int st = 128; st > 0; st >>= 1) {
    if (threadIdx.x < st) sm[threadIdx.x] += sm[threadIdx.x + st];
    __syncthreads();
  }
  if (threadIdx.x == 0) g[col] = sm[0] / (float)kNT;
}

__global__ void k_classifier(const float* __restrict__ g, const float* __restrict__ Wc1,
                             const float* __restrict__ bc1, const float* __restrict__ Wc2,
                             const float* __restrict__ bc2, float* __restrict__ out)
{
  __shared__ float gs[256];
  __shared__ float hr[128];
  int t = threadIdx.x;
  gs[t] = g[t]; gs[t + 128] = g[t + 128];
  __syncthreads();
  float acc = bc1[t];
  for (int k = 0; k < 256; ++k) acc += gs[k] * Wc1[k * 128 + t];
  float h = acc > 0.f ? acc : 0.f;
  hr[t] = h * Wc2[t];
  __syncthreads();
  for (int st = 64; st > 0; st >>= 1) {
    if (t < st) hr[t] += hr[t + st];
    __syncthreads();
  }
  if (t == 0) out[0] = hr[0] + bc2[0];
}

// ---------------------------------------------------------------------------
extern "C" void kernel_launch(void* const* d_in, const int* in_sizes, int n_in,
                              void* d_out, int out_size, void* d_ws, size_t ws_size,
                              hipStream_t stream)
{
  const float* ts_x     = (const float*)d_in[0];
  const float* emb[4]   = { (const float*)d_in[1], (const float*)d_in[2],
                            (const float*)d_in[3], (const float*)d_in[4] };
  const float* Wp_ts    = (const float*)d_in[5];
  const float* bp_ts    = (const float*)d_in[6];
  const float* Wp_c     = (const float*)d_in[7];   // [4,128,256]
  const float* bp_c     = (const float*)d_in[8];   // [4,256]
  const float* Wsrc     = (const float*)d_in[9];   // [L,T,256,256]
  const float* Wdst     = (const float*)d_in[10];
  const float* asrc     = (const float*)d_in[11];  // [L,T,4,64]
  const float* adst     = (const float*)d_in[12];
  const float* bconv    = (const float*)d_in[13];  // [L,T,256]
  const float* Wc1      = (const float*)d_in[14];
  const float* bc1      = (const float*)d_in[15];
  const float* Wc2      = (const float*)d_in[16];
  const float* bc2      = (const float*)d_in[17];
  const int*   ei[8];
  for (int t = 0; t < 8; ++t) ei[t] = (const int*)d_in[18 + t];

  const int szs[5]  = { kNT, kNV, kNDg, kNM, kNP };
  const int off[5]  = { 0, kNT, kNT + kNV, kNT + kNV + kNDg, kNT + kNV + kNDg + kNM };
  const int eoff[4] = { 0, kNV, kNV + kNDg, kNV + kNDg + kNM };  // into emb-h block
  const int esrc[8] = { 0, 1, 0, 2, 0, 3, 0, 4 };
  const int edst[8] = { 1, 0, 2, 0, 3, 0, 4, 0 };

  // ---- workspace partition: f32 region, then f16 region (16B aligned) ----
  float* xcur  = (float*)d_ws;
  float* xnew  = xcur + (size_t)kNTOT * kD;
  float* hsb   = xnew + (size_t)kNTOT * kD;
  float* es    = hsb  + (size_t)kNT * kD;
  float* ed    = es   + (size_t)kNT * kH;
  float* mbuf  = ed   + (size_t)kNT * kH;
  float* den   = mbuf + (size_t)kNT * kH;
  float* exb   = den  + (size_t)kNT * kH;
  float* wdeff = exb  + (size_t)kE * kH;
  float* g     = wdeff + kD * kH;
  _Float16* xh    = (_Float16*)(g + kD);
  _Float16* tsh   = xh   + (size_t)kNTOT * kD;          // ts_x as f16 [NT,64]
  _Float16* embh  = tsh  + (size_t)kNT * 64;            // all embeds f16 [34000,128]
  _Float16* WpTts = embh + (size_t)kNEMB * 128;         // [256,64]
  _Float16* WpTc  = WpTts + 64 * kD;                    // 4 x [256,128]
  _Float16* WsT   = WpTc  + (size_t)4 * 128 * kD;       // 16 x [256,256]

  auto g1 = [](int n) { return dim3((n + 255) / 256); };
  const dim3 gblk(32, 4);

  // ---- one-time (per call) operand preparation: f16 converts + transposes ----
  hipLaunchKernelGGL(k_cvt_h, g1(kNT * 64), 256, 0, stream, ts_x, tsh, kNT * 64);
  for (int t = 0; t < 4; ++t)
    hipLaunchKernelGGL(k_cvt_h, g1(szs[t + 1] * 128), 256, 0, stream,
                       emb[t], embh + (size_t)eoff[t] * 128, szs[t + 1] * 128);
  hipLaunchKernelGGL(k_transpose_h, g1(64 * kD), 256, 0, stream, Wp_ts, WpTts, 64, kD);
  for (int t = 0; t < 4; ++t)
    hipLaunchKernelGGL(k_transpose_h, g1(128 * kD), 256, 0, stream,
                       Wp_c + (size_t)t * 128 * kD, WpTc + (size_t)t * kD * 128, 128, kD);
  for (int lt = 0; lt < kL * kT; ++lt)
    hipLaunchKernelGGL(k_transpose_h, g1(kD * kD), 256, 0, stream,
                       Wsrc + (size_t)lt * kD * kD, WsT + (size_t)lt * kD * kD, kD, kD);

  // ---- node projections -> xcur (f32), then xh (f16) ----
  gemm_wmma_h<64><<<dim3((kNT + 15) / 16), gblk, 0, stream>>>(tsh, WpTts, bp_ts, xcur, kNT);
  for (int t = 1; t < 5; ++t)
    gemm_wmma_h<128><<<dim3((szs[t] + 15) / 16), gblk, 0, stream>>>(
        embh + (size_t)eoff[t - 1] * 128, WpTc + (size_t)(t - 1) * kD * 128,
        bp_c + (size_t)(t - 1) * kD, xcur + (size_t)off[t] * kD, szs[t]);
  hipLaunchKernelGGL(k_cvt_h, g1(kNTOT * kD), 256, 0, stream, xcur, xh, kNTOT * kD);

  // ---- GAT layers ----
  for (int l = 0; l < kL; ++l) {
    hipMemsetAsync(xnew, 0, (size_t)kNTOT * kD * sizeof(float), stream);
    for (int t = 0; t < kT; ++t) {
      const int s = esrc[t], d = edst[t];
      const int Ns = szs[s], Nd = szs[d];
      float* outd = xnew + (size_t)off[d] * kD;
      const size_t lt = (size_t)l * kT + t;
      const float* Wd = Wdst + lt * kD * kD;
      const float* as = asrc + lt * kH * kC;
      const float* ad = adst + lt * kH * kC;
      const float* bc = bconv + lt * kD;

      gemm_wmma_h<256><<<dim3((Ns + 15) / 16), gblk, 0, stream>>>(
          xh + (size_t)off[s] * kD, WsT + lt * kD * kD, nullptr, hsb, Ns);
      hipLaunchKernelGGL(k_wdeff,  g1(kD * kH),  256, 0, stream, Wd, ad, wdeff);
      hipLaunchKernelGGL(k_ed,     g1(Nd * kH),  256, 0, stream,
                         xcur + (size_t)off[d] * kD, wdeff, ed, Nd);
      hipLaunchKernelGGL(k_es,     g1(Ns * kH),  256, 0, stream, hsb, as, es, Ns);
      hipLaunchKernelGGL(k_initmd, g1(Nd * kH),  256, 0, stream, mbuf, den, Nd);
      hipLaunchKernelGGL(k_edgemax, g1(kE * kH), 256, 0, stream, ei[t], es, ed, mbuf);
      hipLaunchKernelGGL(k_edgeexp, g1(kE * kH), 256, 0, stream, ei[t], es, ed, mbuf, den, exb);
      hipLaunchKernelGGL(k_scatter, dim3(kE), dim3(kD), 0, stream, ei[t], hsb, exb, den, outd);
      hipLaunchKernelGGL(k_addbias, g1(Nd * kD), 256, 0, stream, bc, outd, Nd);
    }
    hipLaunchKernelGGL(k_relu_cvt, g1(kNTOT * kD), 256, 0, stream,
                       xnew, xcur, xh, kNTOT * kD);
  }

  // ---- readout + classifier ----
  hipLaunchKernelGGL(k_mean_ts, dim3(kD), dim3(256), 0, stream, xcur, g);
  hipLaunchKernelGGL(k_classifier, dim3(1), dim3(128), 0, stream,
                     g, Wc1, bc1, Wc2, bc2, (float*)d_out);
}
